// LSTMNetwork_69836168233743
// MI455X (gfx1250) — matrix operand order repaired and checked by reading, not compiled, not run on hardware
//
#include <hip/hip_runtime.h>

// ---------------------------------------------------------------------------
// LSTM (B=16384, T=512, I=3, H=32, O=1) for MI455X / gfx1250, wave32 + WMMA.
//
// One wave handles 16 batch rows. Per timestep:
//   acc[n]  = wmma_f32_16x16x4_f32 (A = x_t padded 3->4, B = W_ih tile, C = bias)
//   acc[n] += wmma_f32_16x16x32_f16(A = h (16x32 f16),   B = W_hh tile)  (8 tiles, 4H=128)
// Gate tiles: 0-1 = i, 2-3 = f, 4-5 = g, 6-7 = o.
// The C/D layout is the transpose of the A layout, so h round-trips through a
// per-wave 1KB LDS tile using a common K-permutation (0,16,1,17,...,15,31)
// applied to BOTH the h fragment and the W_hh fragment (GEMM-invariant),
// making the store a single packed b32 per row and the reload 2x ds_load_b128.
// ---------------------------------------------------------------------------

typedef __attribute__((ext_vector_type(2)))  float    v2f;
typedef __attribute__((ext_vector_type(8)))  float    v8f;
typedef __attribute__((ext_vector_type(2)))  _Float16 v2h;
typedef __attribute__((ext_vector_type(8)))  _Float16 v8h;
typedef __attribute__((ext_vector_type(16))) _Float16 v16h;
typedef __attribute__((ext_vector_type(4)))  unsigned v4u;

#define B_SZ 16384
#define T_SZ 512
#define I_SZ 3
#define H_SZ 32
#define NT   8            // 8 N-tiles of 16 cover 4H = 128 gate outputs
#define WAVES_PER_BLOCK 8
#define ROWS_PER_WAVE   16

__device__ __forceinline__ float sigm(float x)  { return 1.0f / (1.0f + __expf(-x)); }
__device__ __forceinline__ float tanhfa(float x){ return 2.0f / (1.0f + __expf(-2.0f * x)) - 1.0f; }

__global__ __launch_bounds__(256, 1)
void lstm_wmma_gfx1250(const float* __restrict__ x,
                       const float* __restrict__ W_ih,
                       const float* __restrict__ W_hh,
                       const float* __restrict__ b_ih,
                       const float* __restrict__ b_hh,
                       const float* __restrict__ W_fc,
                       const float* __restrict__ b_fc,
                       float* __restrict__ out)
{
    // Per-wave private h staging tile: 16 rows x 32 halfs (stored as packed pairs).
    __shared__ unsigned hbuf[WAVES_PER_BLOCK][ROWS_PER_WAVE * 16];

    const int  tid    = threadIdx.x;
    const int  wave   = tid >> 5;
    const int  lane   = tid & 31;
    const int  c16    = lane & 15;       // column within a 16-wide tile / A row id
    const bool hiHalf = lane >= 16;

    const int rowBase = blockIdx.x * (WAVES_PER_BLOCK * ROWS_PER_WAVE) + wave * ROWS_PER_WAVE;

    // ---- loop-invariant fragments -----------------------------------------
    // W_hh B-fragment, K-permuted: lane<16 holds N=16n+c16, pairs (k, k+16) k=0..7;
    // lane>=16 holds same N, pairs (k, k+16) k=8..15.
    v16h  Bhh[NT];
    float bi0[NT], bi1[NT];   // W_ih B-fragment for the 16x16x4 f32 WMMA
    float biasv[NT];          // b_ih + b_hh per output column
    {
        const int jbase = hiHalf ? 8 : 0;
#pragma unroll
        for (int n = 0; n < NT; ++n) {
            const int   N  = 16 * n + c16;
            const float* wr = W_hh + (size_t)N * H_SZ;
            v16h bf;
#pragma unroll
            for (int j = 0; j < 8; ++j) {
                const int jj  = jbase + j;
                bf[2 * j]     = (_Float16)wr[jj];
                bf[2 * j + 1] = (_Float16)wr[jj + 16];
            }
            Bhh[n] = bf;

            const float* wi = W_ih + (size_t)N * I_SZ;
            bi0[n] = hiHalf ? wi[2] : wi[0];    // K=2 : K=0
            bi1[n] = hiHalf ? 0.0f  : wi[1];    // K=3(pad) : K=1
            biasv[n] = b_ih[N] + b_hh[N];
        }
    }

    // ---- state -------------------------------------------------------------
    float cst[16];                       // cell state: [pair p][vgpr v] -> p*8+v
#pragma unroll
    for (int k = 0; k < 16; ++k) cst[k] = 0.0f;
    v16h hA = (v16h)(_Float16)0.0f;      // h fragment (A layout), h0 = 0

    const float* xrow = x + (size_t)(rowBase + c16) * T_SZ * I_SZ;
    const v4u*   h4   = (const v4u*)&hbuf[wave][0];   // 4x b128 granules per row

    // ---- recurrent scan ----------------------------------------------------
#pragma unroll 1
    for (int t = 0; t < T_SZ; ++t) {
        // x_t A-fragment (16x4 f32, K=3 padded with 0)
        const float* xp = xrow + (size_t)t * I_SZ;
        if (t + 16 < T_SZ) __builtin_prefetch(xp + 16 * I_SZ, 0, 0);
        float a0 = hiHalf ? xp[2] : xp[0];
        float a1 = hiHalf ? 0.0f  : xp[1];
        v2f ax = {a0, a1};

        v8f acc[NT];
#pragma unroll
        for (int n = 0; n < NT; ++n) {
            const float b = biasv[n];
            v8f cinit = {b, b, b, b, b, b, b, b};
            v2f bx = {bi0[n], bi1[n]};
            // gates = bias + x_t @ W_ih^T ...
            acc[n] = __builtin_amdgcn_wmma_f32_16x16x4_f32(
                false, ax, false, bx, (short)0, cinit, false, false);
            // ... + h_{t-1} @ W_hh^T
            acc[n] = __builtin_amdgcn_wmma_f32_16x16x32_f16(
                false, hA, false, Bhh[n], (short)0, acc[n], false, false);
        }

        // element-wise cell update; row m = v + 8*hiHalf, cols c16 and c16+16
#pragma unroll
        for (int v = 0; v < 8; ++v) {
            v2h hp;
#pragma unroll
            for (int p = 0; p < 2; ++p) {
                const float ig = sigm(acc[0 + p][v]);
                const float fg = sigm(acc[2 + p][v]);
                const float gg = tanhfa(acc[4 + p][v]);
                const float og = sigm(acc[6 + p][v]);
                const float cn = fg * cst[p * 8 + v] + ig * gg;
                cst[p * 8 + v] = cn;
                hp[p] = (_Float16)(og * tanhfa(cn));
            }
            const int m = v + (hiHalf ? 8 : 0);
            hbuf[wave][m * 16 + c16] = __builtin_bit_cast(unsigned, hp); // ds_store_b32
        }

        // reload h in A layout (K-permuted): 2x ds_load_b128; per-wave LDS is
        // private and same-wave DS ops are in-order, so no barrier needed.
        {
            const int m = c16;
            v4u lov = h4[m * 4 + (hiHalf ? 2 : 0)];
            v4u hiv = h4[m * 4 + (hiHalf ? 3 : 1)];
            v8h lo = __builtin_bit_cast(v8h, lov);
            v8h hi = __builtin_bit_cast(v8h, hiv);
            hA = __builtin_shufflevector(lo, hi, 0, 1, 2, 3, 4, 5, 6, 7,
                                                 8, 9, 10, 11, 12, 13, 14, 15);
        }
    }

    // ---- final FC: out = sigmoid(h_T @ W_fc^T + b_fc), O = 1 ---------------
    if (lane < 16) {
        const _Float16* hh = (const _Float16*)&hbuf[wave][0];
        float s = b_fc[0];
#pragma unroll
        for (int j = 0; j < H_SZ; ++j) {
            // h[m][j] lives at half-index m*32 + (j%16)*2 + (j/16)  (K-permuted)
            const float hv = (float)hh[lane * 32 + (j & 15) * 2 + (j >> 4)];
            s += hv * W_fc[j];
        }
        out[rowBase + lane] = sigm(s);
    }
}

extern "C" void kernel_launch(void* const* d_in, const int* in_sizes, int n_in,
                              void* d_out, int out_size, void* d_ws, size_t ws_size,
                              hipStream_t stream) {
    (void)in_sizes; (void)n_in; (void)d_ws; (void)ws_size; (void)out_size;
    const float* x    = (const float*)d_in[0];
    const float* W_ih = (const float*)d_in[1];
    const float* W_hh = (const float*)d_in[2];
    const float* b_ih = (const float*)d_in[3];
    const float* b_hh = (const float*)d_in[4];
    const float* W_fc = (const float*)d_in[5];
    const float* b_fc = (const float*)d_in[6];
    float* out = (float*)d_out;

    const int rowsPerBlock = WAVES_PER_BLOCK * ROWS_PER_WAVE;   // 128
    dim3 grid(B_SZ / rowsPerBlock);                             // 128 blocks
    dim3 block(256);                                            // 8 waves
    lstm_wmma_gfx1250<<<grid, block, 0, stream>>>(x, W_ih, W_hh, b_ih, b_hh,
                                                  W_fc, b_fc, out);
}
